// _CustomLSTM_4337916969415
// MI455X (gfx1250) — matrix operand order repaired and checked by reading, not compiled, not run on hardware
//
#include <hip/hip_runtime.h>
#include <hip/hip_bf16.h>
#include <math.h>

// ---------------------------------------------------------------------------
// CDNA5 WMMA fragment helpers (wave32, V_WMMA_F32_16X16X32_BF16)
// ---------------------------------------------------------------------------
typedef __attribute__((ext_vector_type(16))) __bf16 v16bf;
typedef __attribute__((ext_vector_type(8)))  float  v8f;

union FragU { v16bf v; uint4 u[2]; };

// A-matrix 16x32 bf16, source row-major [M][K] (pitch in elements).
__device__ __forceinline__ v16bf frag_a(const __bf16* base, int pitch, int lr, int lh) {
  const __bf16* p = base + (size_t)lr * pitch + lh * 8;
  FragU f;
  f.u[0] = *(const uint4*)(p);        // K = lh*8 + 0..7
  f.u[1] = *(const uint4*)(p + 16);   // K = lh*8 + 16..23
  return f.v;
}

// B-matrix 32x16 bf16, source row-major [N][K] (mathematical B = W^T).
__device__ __forceinline__ v16bf frag_b(const __bf16* base, int pitch, int lr, int lh) {
  const __bf16* p = base + (size_t)lr * pitch + lh * 8;
  FragU f;
  f.u[0] = *(const uint4*)(p);        // K = lh*8 + 0..7
  f.u[1] = *(const uint4*)(p + 16);   // K = lh*8 + 16..23
  return f.v;
}

__device__ __forceinline__ v8f wmma_bf16(v16bf a, v16bf b, v8f c) {
  return __builtin_amdgcn_wmma_f32_16x16x32_bf16(false, a, false, b, (short)0, c,
                                                 false, false);
}

__device__ __forceinline__ float sigmf(float x) { return 1.0f / (1.0f + expf(-x)); }

// ---- CDNA5 async global->LDS copy (ASYNCcnt path, bypasses VGPRs) ----------
__device__ __forceinline__ void async_copy16(const void* gsrc, void* ldst) {
  unsigned lds = (unsigned)(unsigned long long)ldst;    // low 32b = LDS offset
  unsigned long long g = (unsigned long long)gsrc;
  asm volatile("global_load_async_to_lds_b128 %0, %1, off"
               :: "v"(lds), "v"(g) : "memory");
}

__device__ __forceinline__ void wait_async0() {
#if __has_builtin(__builtin_amdgcn_s_wait_asynccnt)
  __builtin_amdgcn_s_wait_asynccnt(0);
#else
  asm volatile("s_wait_asynccnt 0x0" ::: "memory");
#endif
}

// ---------------------------------------------------------------------------
// Kernel 0: f32 -> bf16 conversion (16B-vectorized both ways)
// ---------------------------------------------------------------------------
__global__ __launch_bounds__(256) void k_f2bf(const float4* __restrict__ src,
                                              uint4* __restrict__ dst, int n8) {
  int i = blockIdx.x * 256 + threadIdx.x;     // one uint4 (8 bf16) per thread
  if (i < n8) {
    float4 a = src[2 * i], b = src[2 * i + 1];
    union { __bf16 h[8]; uint4 u; } p;
    p.h[0] = (__bf16)a.x; p.h[1] = (__bf16)a.y; p.h[2] = (__bf16)a.z; p.h[3] = (__bf16)a.w;
    p.h[4] = (__bf16)b.x; p.h[5] = (__bf16)b.y; p.h[6] = (__bf16)b.z; p.h[7] = (__bf16)b.w;
    dst[i] = p.u;
  }
}

// ---------------------------------------------------------------------------
// Kernel 1: gated = input @ Wx^T   (M=16384, N=4096, K=1024), bf16 operands.
// 128x128 tile per WG, double-buffered LDS filled with async-to-LDS copies.
// ---------------------------------------------------------------------------
__global__ __launch_bounds__(256) void k_gemm_x(const __bf16* __restrict__ A,
                                                const __bf16* __restrict__ W,
                                                float* __restrict__ C,
                                                int M, int N, int K) {
  const int AP = 40;                     // LDS pitch (bf16), bank-friendly
  __shared__ __bf16 As[2][128 * 40];
  __shared__ __bf16 Bs[2][128 * 40];

  const int m0 = blockIdx.y * 128, n0 = blockIdx.x * 128;
  const int t = threadIdx.x;
  const int w = t >> 5;
  const int wm = w & 1, wn = w >> 1;     // 2 (M) x 4 (N) wave grid
  const int l = t & 31, lr = l & 15, lh = l >> 4;

  v8f acc[4][2];
  for (int i = 0; i < 4; i++)
    for (int j = 0; j < 2; j++) acc[i][j] = v8f{};

  // issue async copies of one 128x32 bf16 tile pair (16B chunks, 4/thread)
  auto stage = [&](int buf, int k0) {
    for (int i = 0; i < 2; i++) {
      int c = t + i * 256;               // chunk 0..511
      int row = c >> 2, col8 = (c & 3) * 8;
      async_copy16(A + (size_t)(m0 + row) * K + k0 + col8,
                   &As[buf][row * AP + col8]);
      async_copy16(W + (size_t)(n0 + row) * K + k0 + col8,
                   &Bs[buf][row * AP + col8]);
    }
  };

  stage(0, 0);
  wait_async0();
  __syncthreads();

  for (int k0 = 0; k0 < K; k0 += 32) {
    int buf = (k0 >> 5) & 1;
    if (k0 + 32 < K) stage(buf ^ 1, k0 + 32);   // overlap with WMMA below

    v16bf bf[2], af[4];
    for (int j = 0; j < 2; j++) bf[j] = frag_b(&Bs[buf][(wn * 32 + j * 16) * AP], AP, lr, lh);
    for (int i = 0; i < 4; i++) af[i] = frag_a(&As[buf][(wm * 64 + i * 16) * AP], AP, lr, lh);
    for (int i = 0; i < 4; i++)
      for (int j = 0; j < 2; j++)
        acc[i][j] = wmma_bf16(af[i], bf[j], acc[i][j]);

    wait_async0();
    __syncthreads();
  }

  for (int i = 0; i < 4; i++)
    for (int j = 0; j < 2; j++) {
      int mb = m0 + wm * 64 + i * 16 + lh * 8;
      int nb = n0 + wn * 32 + j * 16 + lr;
      for (int v = 0; v < 8; v++)
        C[(size_t)(mb + v) * N + nb] = acc[i][j][v];
    }
}

// ---------------------------------------------------------------------------
// Kernel 2: persistent LayerNorm-LSTM scan.
// 32 WGs x 256 threads; WG owns j-slice [32*wg, 32*wg+32) of H and the 4
// matching gate slices (128 gate rows). Wh slice resident in LDS as bf16
// (128 x 1032 = 258 KB -> needs CDNA5's 320 KB WGP LDS).
// Cross-WG layernorm stats via global_atomic_add_f32 + grid barrier.
// ---------------------------------------------------------------------------
#define NWG 32

__global__ __launch_bounds__(256) void k_lstm(
    const float* __restrict__ gated,   // [T*B][4H]
    const float* __restrict__ h0,      // [B][H]
    const float* __restrict__ c0,      // [B][H]
    const float* __restrict__ Wh,      // [4H][H] f32
    const float* __restrict__ gn_g, const float* __restrict__ gn_b,
    const float* __restrict__ cn_g, const float* __restrict__ cn_b,
    __bf16* __restrict__ hbuf,         // [B][H] bf16 (shared hidden state)
    float* __restrict__ stats,         // [T][B][4] (gsum,gsq,csum,csq), zeroed
    int* __restrict__ bar,             // grid barrier counter, zeroed
    float* __restrict__ out)           // output[T*B*H] ++ h[B*H] ++ c[B*H]
{
  const int T = 512, B = 32, H = 1024, G = 4096;
  const int WP = 1032;                  // Wh LDS pitch (bf16)
  const int GP = 132;                   // gates LDS pitch (f32)
  extern __shared__ char smem[];
  __bf16* WhL    = (__bf16*)smem;                      // 128 x 1032
  float*  gatesL = (float*)(smem + 128 * WP * 2);      // 32 x 132

  const int wg = blockIdx.x;
  const int j0 = wg * 32;
  const int t  = threadIdx.x;

  // ---- load this WG's Wh slice into LDS (f32 -> bf16, one time) ----
  for (int e = t; e < 128 * (H / 4); e += 256) {
    int r = e >> 8;                     // local gate row 0..127
    int c4 = (e & 255) * 4;
    int grow = (r >> 5) * H + j0 + (r & 31);
    float4 v = *(const float4*)(Wh + (size_t)grow * H + c4);
    __bf16* d = &WhL[r * WP + c4];
    d[0] = (__bf16)v.x; d[1] = (__bf16)v.y; d[2] = (__bf16)v.z; d[3] = (__bf16)v.w;
  }

  // ---- per-thread cell state: 4 cells (b, jl), same b across q ----
  const int b = t & 31;
  const int jbase = t >> 5;             // 0..7
  float c_reg[4], h_reg[4];
  for (int q = 0; q < 4; q++) {
    int j = j0 + jbase + 8 * q;
    c_reg[q] = c0[(size_t)b * H + j];
    float hv = h0[(size_t)b * H + j];
    h_reg[q] = hv;
    hbuf[(size_t)b * H + j] = (__bf16)hv;
  }

  int barid = 0;
  auto gbar = [&]() {
    barid++;
    __syncthreads();
    if (t == 0) {
      __threadfence();
      atomicAdd(bar, 1);
      while (__hip_atomic_load(bar, __ATOMIC_RELAXED, __HIP_MEMORY_SCOPE_AGENT)
             < barid * NWG) {
        __builtin_amdgcn_s_sleep(1);
      }
    }
    __syncthreads();
    __threadfence();
  };
  gbar();                               // barrier 0: hbuf + WhL ready

  const int wv = t >> 5;                // wave 0..7
  const int l = t & 31, lr = l & 15, lh = l >> 4;
  const int n0 = wv * 16;               // wave's n-tile within the 128 cols

  for (int step = 0; step < T; step++) {
    // prefetch next step's input-projection block into cache
    if (step + 1 < T) {
      const float* nx = gated + (size_t)(step + 1) * B * G;
      int pm = t & 31, ps = (t >> 5) & 3;        // 32 rows x 4 gate chunks
      __builtin_prefetch(nx + (size_t)pm * G + ps * H + j0, 0, 3);
    }

    // ---- Phase A: gates[32 x 128] = h @ WhL^T via WMMA ----
    v8f acc0 = v8f{}, acc1 = v8f{};
    for (int k0 = 0; k0 < H; k0 += 32) {
      v16bf a0 = frag_a(hbuf + k0, H, lr, lh);                  // rows 0..15
      v16bf a1 = frag_a(hbuf + (size_t)16 * H + k0, H, lr, lh); // rows 16..31
      v16bf bw = frag_b(&WhL[n0 * WP + k0], WP, lr, lh);
      acc0 = wmma_bf16(a0, bw, acc0);
      acc1 = wmma_bf16(a1, bw, acc1);
    }
    // add precomputed input projection, scatter to LDS
    const float* gx = gated + (size_t)step * B * G;
    {
      int n = n0 + lr;
      int grow = (n >> 5) * H + j0 + (n & 31);
      for (int v = 0; v < 8; v++) {
        int m = v + lh * 8;
        gatesL[m * GP + n]        = acc0[v] + gx[(size_t)m * G + grow];
        gatesL[(m + 16) * GP + n] = acc1[v] + gx[(size_t)(m + 16) * G + grow];
      }
    }
    __syncthreads();

    // ---- gate-layernorm partial stats (this WG's 128 of 4096 cols) ----
    if (t < 32) {
      float s = 0.f, s2 = 0.f;
      for (int n = 0; n < 128; n++) {
        float x = gatesL[t * GP + n];
        s += x; s2 += x * x;
      }
      atomicAdd(&stats[((size_t)step * B + t) * 4 + 0], s);
      atomicAdd(&stats[((size_t)step * B + t) * 4 + 1], s2);
    }
    gbar();                             // barrier 1: gate stats complete

    float gs  = stats[((size_t)step * B + b) * 4 + 0];
    float gs2 = stats[((size_t)step * B + b) * 4 + 1];
    float mu  = gs * (1.0f / 4096.0f);
    float var = gs2 * (1.0f / 4096.0f) - mu * mu;
    float rs  = rsqrtf(var + 1e-5f);

    float csum = 0.f, csum2 = 0.f;
    for (int q = 0; q < 4; q++) {
      int jl = jbase + 8 * q;
      int j  = j0 + jl;
      float iv = (gatesL[b * GP + jl     ] - mu) * rs * gn_g[j      ] + gn_b[j      ];
      float fv = (gatesL[b * GP + 32 + jl] - mu) * rs * gn_g[H + j  ] + gn_b[H + j  ];
      float gv = (gatesL[b * GP + 64 + jl] - mu) * rs * gn_g[2*H + j] + gn_b[2*H + j];
      float cn = sigmf(fv) * c_reg[q] + sigmf(iv) * tanhf(gv);
      c_reg[q] = cn;                    // pre-norm cell
      csum += cn; csum2 += cn * cn;
    }
    atomicAdd(&stats[((size_t)step * B + b) * 4 + 2], csum);
    atomicAdd(&stats[((size_t)step * B + b) * 4 + 3], csum2);
    gbar();                             // barrier 2: cell stats complete

    float cs   = stats[((size_t)step * B + b) * 4 + 2];
    float cs2  = stats[((size_t)step * B + b) * 4 + 3];
    float muc  = cs * (1.0f / 1024.0f);
    float varc = cs2 * (1.0f / 1024.0f) - muc * muc;
    float rsc  = rsqrtf(varc + 1e-5f);

    for (int q = 0; q < 4; q++) {
      int jl = jbase + 8 * q;
      int j  = j0 + jl;
      float cn = (c_reg[q] - muc) * rsc * cn_g[j] + cn_b[j];
      float ov = (gatesL[b * GP + 96 + jl] - mu) * rs * gn_g[3*H + j] + gn_b[3*H + j];
      float hv = sigmf(ov) * tanhf(cn);
      c_reg[q] = cn;
      h_reg[q] = hv;
      hbuf[(size_t)b * H + j] = (__bf16)hv;
      out[((size_t)step * B + b) * H + j] = hv;
    }
    gbar();                             // barrier 3: new h visible to all WGs
  }

  // final h and c segments of d_out
  const size_t OH = (size_t)T * B * H;
  for (int q = 0; q < 4; q++) {
    int j = j0 + jbase + 8 * q;
    out[OH + (size_t)b * H + j]                 = h_reg[q];
    out[OH + (size_t)B * H + (size_t)b * H + j] = c_reg[q];
  }
}

// ---------------------------------------------------------------------------
extern "C" void kernel_launch(void* const* d_in, const int* in_sizes, int n_in,
                              void* d_out, int out_size, void* d_ws, size_t ws_size,
                              hipStream_t stream) {
  (void)in_sizes; (void)n_in; (void)out_size; (void)ws_size;
  const float* input = (const float*)d_in[0];
  const float* h0    = (const float*)d_in[1];
  const float* c0    = (const float*)d_in[2];
  const float* Wx    = (const float*)d_in[3];
  const float* Wh    = (const float*)d_in[4];
  const float* gn_g  = (const float*)d_in[5];
  const float* gn_b  = (const float*)d_in[6];
  const float* cn_g  = (const float*)d_in[7];
  const float* cn_b  = (const float*)d_in[8];
  float* out = (float*)d_out;

  const int T = 512, B = 32, D = 1024, H = 1024, G = 4096;

  char* ws = (char*)d_ws;
  size_t off = 0;
  float*  gated = (float*)(ws + off);  off += (size_t)T * B * G * sizeof(float); // 256 MB
  __bf16* hbuf  = (__bf16*)(ws + off); off += (size_t)B * H * sizeof(__bf16);    // 64 KB
  float*  stats = (float*)(ws + off);
  size_t  statsB = (size_t)T * B * 4 * sizeof(float);                            // 256 KB
  int*    bar   = (int*)(ws + off + statsB);
  size_t  zoff  = off;                 off += statsB + 256;
  __bf16* Abf   = (__bf16*)(ws + off); off += (size_t)T * B * D * sizeof(__bf16); // 32 MB
  __bf16* Wxbf  = (__bf16*)(ws + off); off += (size_t)G * D * sizeof(__bf16);     //  8 MB

  // zero stats accumulators + barrier counter (replayed every graph launch)
  hipMemsetAsync(ws + zoff, 0, statsB + 256, stream);

  // Phase 0: one-time f32 -> bf16 conversion of GEMM operands
  int nA8 = (T * B * D) / 8, nW8 = (G * D) / 8;
  k_f2bf<<<dim3((nA8 + 255) / 256), 256, 0, stream>>>((const float4*)input, (uint4*)Abf, nA8);
  k_f2bf<<<dim3((nW8 + 255) / 256), 256, 0, stream>>>((const float4*)Wx, (uint4*)Wxbf, nW8);

  // Phase 1: input projection (async-to-LDS double-buffered WMMA GEMM)
  dim3 g1(G / 128, (T * B) / 128);
  k_gemm_x<<<g1, 256, 0, stream>>>(Abf, Wxbf, gated, T * B, G, D);

  // Phase 2: persistent recurrent scan, 32 WGs, 258 KB Wh + 17 KB gates in LDS
  size_t lds = (size_t)128 * 1032 * sizeof(__bf16) + (size_t)32 * 132 * sizeof(float);
  hipFuncSetAttribute(reinterpret_cast<const void*>(&k_lstm),
                      hipFuncAttributeMaxDynamicSharedMemorySize, (int)lds);
  k_lstm<<<dim3(NWG), 256, lds, stream>>>(gated, h0, c0, Wh,
                                          gn_g, gn_b, cn_g, cn_b,
                                          hbuf, stats, bar, out);
}
